// FadeFeatureAlignment_11819749998712
// MI455X (gfx1250) — matrix operand order repaired and checked by reading, not compiled, not run on hardware
//
#include <hip/hip_runtime.h>
#include <hip/hip_bf16.h>
#include <math.h>

typedef __attribute__((ext_vector_type(16))) __bf16 v16bf;
typedef __attribute__((ext_vector_type(8)))  __bf16 v8bf;
typedef __attribute__((ext_vector_type(8)))  float  v8f;

#define WAVES 8   // 256 threads = 8 wave32 per block

static __device__ __forceinline__ v16bf cat8(v8bf lo, v8bf hi) {
  return __builtin_shufflevector(lo, hi, 0, 1, 2, 3, 4, 5, 6, 7,
                                 8, 9, 10, 11, 12, 13, 14, 15);
}

// ---------------------------------------------------------------------------
// Pack fp32 weight W[K,Nsrc] into WMMA-B fragment order (bf16), N padded to
// Npad with zeros.  For each 32(K)x16(N) tile: lane = (k%16)/8*16 + n%16,
// t = k%8 + (k%32)/16*8  ->  dst[((kt*(Npad/16)+nt)*32 + lane)*16 + t]
// ---------------------------------------------------------------------------
__global__ void k_pack_b(const float* __restrict__ W, __bf16* __restrict__ dst,
                         int K, int Nsrc, int Npad) {
  int e = blockIdx.x * blockDim.x + threadIdx.x;
  if (e >= K * Npad) return;
  int k = e / Npad, n = e % Npad;
  float v = (n < Nsrc) ? W[(size_t)k * Nsrc + n] : 0.0f;
  int kt = k >> 5, kr = k & 31, kh = (kr >> 3) & 1, t = (kr & 7) | ((kr >> 4) << 3);
  int nt = n >> 4, nl = n & 15, lane = kh * 16 + nl;
  int ntn = Npad >> 4;
  dst[(((size_t)kt * ntn + nt) * 32 + lane) * 16 + t] = (__bf16)v;
}

// ---------------------------------------------------------------------------
// gate = sigmoid(coarse . gate_w + gate_b) at coarse resolution.
// ---------------------------------------------------------------------------
__global__ void k_gate(const float* __restrict__ coarse, const float* __restrict__ gw,
                       const float* __restrict__ gb, float* __restrict__ gate, int npix) {
  int lane = threadIdx.x & 31;
  int p = blockIdx.x * (blockDim.x >> 5) + (threadIdx.x >> 5);
  if (p >= npix) return;
  const float* row = coarse + (size_t)p * 256;
  float acc = 0.f;
#pragma unroll
  for (int u = 0; u < 8; ++u) {
    int c = lane * 8 + u;
    acc += row[c] * gw[c];
  }
  for (int m = 16; m >= 1; m >>= 1) acc += __shfl_xor(acc, m, 32);
  if (lane == 0) gate[p] = 1.0f / (1.0f + __expf(-(acc + gb[0])));
}

// ---------------------------------------------------------------------------
// Cooperative copy of the block's packed-B slab (NB n-tiles x KT k-tiles)
// into LDS, preserving fragment order: sB[((kt*NB + u)*32 + lane)*16 + t]
// ---------------------------------------------------------------------------
template <int KT, int NB>
static __device__ __forceinline__ void stage_b(const __bf16* __restrict__ Bp,
                                               __bf16* __restrict__ sB,
                                               int ntn, int nt0) {
  for (int c = threadIdx.x; c < KT * NB * 32; c += 256) {
    int kt = c / (NB * 32);
    int rr = c % (NB * 32);
    int u = rr / 32, lc = rr % 32;
    const __bf16* src = Bp + (((size_t)kt * ntn + nt0 + u) * 32 + lc) * 16;
    __bf16* dst = sB + ((size_t)c) * 16;
    *(v8bf*)dst = *(const v8bf*)src;
    *(v8bf*)(dst + 8) = *(const v8bf*)(src + 8);
  }
  __syncthreads();
}

// ---------------------------------------------------------------------------
// GEMM, fp32 A (cvt bf16 on load), LDS-staged B, bf16 D = A*B + bias.
// Each wave computes a 32 x (16*NB) tile: MB=2 m-tiles x NB n-tiles,
// 2*NB WMMAs per K-step.  grid.x = M/(32*WAVES), grid.y = N/16/NB.
// If PH>0 (with PW): store into zero-halo padded NHWC layout
// [img][(PH+2)][(PW+2)][N]; linear->padded math uses compile-time shifts.
// ---------------------------------------------------------------------------
template <int KT, int NB, int PH, int PW>
__global__ void k_gemm_a32(const float* __restrict__ A, const __bf16* __restrict__ Bp,
                           const float* __restrict__ bias, __bf16* __restrict__ D,
                           int N) {
  const int K = KT * 32;
  __shared__ __bf16 sB[KT * NB * 32 * 16];
  int lane = threadIdx.x & 31;
  int wave = threadIdx.x >> 5;
  int nt0 = blockIdx.y * NB;
  stage_b<KT, NB>(Bp, sB, N >> 4, nt0);

  int m0 = (blockIdx.x * WAVES + wave) * 32;
  int hi = lane >> 4;
  v8f acc[2][NB] = {};
  for (int kt = 0; kt < KT; ++kt) {
    v16bf a[2];
#pragma unroll
    for (int mm = 0; mm < 2; ++mm) {
      int row = m0 + mm * 16 + (lane & 15);
      const float* ap = A + (size_t)row * K + kt * 32 + hi * 8;
      if (kt + 1 < KT) __builtin_prefetch(ap + 32, 0, 0);
#pragma unroll
      for (int t = 0; t < 8; ++t) {
        a[mm][t]     = (__bf16)ap[t];
        a[mm][t + 8] = (__bf16)ap[t + 16];
      }
    }
#pragma unroll
    for (int u = 0; u < NB; ++u) {
      v16bf b = *(const v16bf*)(sB + (((kt * NB + u) * 32 + lane) * 16));
#pragma unroll
      for (int mm = 0; mm < 2; ++mm)
        acc[mm][u] = __builtin_amdgcn_wmma_f32_16x16x32_bf16(false, a[mm], false, b,
                                                             (short)0, acc[mm][u],
                                                             false, false);
    }
  }
#pragma unroll
  for (int u = 0; u < NB; ++u) {
    int col = (nt0 + u) * 16 + (lane & 15);
    float bv = bias ? bias[col] : 0.0f;
#pragma unroll
    for (int mm = 0; mm < 2; ++mm)
#pragma unroll
      for (int v = 0; v < 8; ++v) {
        int r = m0 + mm * 16 + v + hi * 8;
        size_t drow;
        if constexpr (PH > 0) {
          int img = r / (PH * PW);          // compile-time pow2 -> shifts
          int rem = r % (PH * PW);
          int i = rem / PW, j = rem % PW;
          drow = (size_t)img * (PH + 2) * (PW + 2) + (size_t)(i + 1) * (PW + 2) + (j + 1);
        } else {
          drow = (size_t)r;
        }
        D[drow * N + col] = (__bf16)(acc[mm][u][v] + bv);
      }
  }
}

// ---------------------------------------------------------------------------
// Same register/LDS-blocked GEMM with bf16 A (unpadded store).
// ---------------------------------------------------------------------------
template <int KT, int NB>
__global__ void k_gemm_a16(const __bf16* __restrict__ A, const __bf16* __restrict__ Bp,
                           const float* __restrict__ bias, __bf16* __restrict__ D,
                           int N) {
  const int K = KT * 32;
  __shared__ __bf16 sB[KT * NB * 32 * 16];
  int lane = threadIdx.x & 31;
  int wave = threadIdx.x >> 5;
  int nt0 = blockIdx.y * NB;
  stage_b<KT, NB>(Bp, sB, N >> 4, nt0);

  int m0 = (blockIdx.x * WAVES + wave) * 32;
  int hi = lane >> 4;
  v8f acc[2][NB] = {};
  for (int kt = 0; kt < KT; ++kt) {
    v16bf a[2];
#pragma unroll
    for (int mm = 0; mm < 2; ++mm) {
      int row = m0 + mm * 16 + (lane & 15);
      const __bf16* ap = A + (size_t)row * K + kt * 32 + hi * 8;
      if (kt + 1 < KT) __builtin_prefetch(ap + 32, 0, 0);
      a[mm] = cat8(*(const v8bf*)ap, *(const v8bf*)(ap + 16));
    }
#pragma unroll
    for (int u = 0; u < NB; ++u) {
      v16bf b = *(const v16bf*)(sB + (((kt * NB + u) * 32 + lane) * 16));
#pragma unroll
      for (int mm = 0; mm < 2; ++mm)
        acc[mm][u] = __builtin_amdgcn_wmma_f32_16x16x32_bf16(false, a[mm], false, b,
                                                             (short)0, acc[mm][u],
                                                             false, false);
    }
  }
#pragma unroll
  for (int u = 0; u < NB; ++u) {
    int col = (nt0 + u) * 16 + (lane & 15);
    float bv = bias ? bias[col] : 0.0f;
#pragma unroll
    for (int mm = 0; mm < 2; ++mm)
#pragma unroll
      for (int v = 0; v < 8; ++v) {
        int r = m0 + mm * 16 + v + hi * 8;
        D[(size_t)r * N + col] = (__bf16)(acc[mm][u][v] + bv);
      }
  }
}

// ---------------------------------------------------------------------------
// 3x3 SAME conv E=64 -> 25 (padded N=32) as 9 shifted WMMA GEMMs over a
// ZERO-HALO padded embedding [img][(HH+2)][(WW+2)][64]: no bounds checks,
// no exec masking -- every shift is an immediate offset off one base pointer.
// MB=2 pixel-tiles x NB=2 n-tiles, all loops unrolled, weights in LDS.
// Writes fp32 logits [npix, 25] (unpadded).
// ---------------------------------------------------------------------------
template <int HH, int WW>
__global__ void k_conv3x3(const __bf16* __restrict__ embp, const __bf16* __restrict__ Bp,
                          const float* __restrict__ bias, float* __restrict__ out) {
  const int PW = WW + 2;
  __shared__ __bf16 sB[18 * 2 * 32 * 16];
  int lane = threadIdx.x & 31;
  int wave = threadIdx.x >> 5;
  stage_b<18, 2>(Bp, sB, 2, 0);

  int m0 = (blockIdx.x * WAVES + wave) * 32;
  int hi = lane >> 4;
  const __bf16* base[2];
#pragma unroll
  for (int mm = 0; mm < 2; ++mm) {
    int p = m0 + mm * 16 + (lane & 15);
    int img = p / (HH * WW);              // pow2 -> shifts
    int rem = p % (HH * WW);
    int i = rem / WW, j = rem % WW;
    base[mm] = embp + ((size_t)img * (HH + 2) * PW + (size_t)(i + 1) * PW + (j + 1)) * 64;
  }
  v8f acc[2][2] = {};
#pragma unroll
  for (int dy = -1; dy <= 1; ++dy) {
#pragma unroll
    for (int dx = -1; dx <= 1; ++dx) {
      int s = (dy + 1) * 3 + (dx + 1);
#pragma unroll
      for (int kk = 0; kk < 2; ++kk) {
        int off = (dy * PW + dx) * 64 + kk * 32 + hi * 8;
        v16bf a[2];
#pragma unroll
        for (int mm = 0; mm < 2; ++mm)
          a[mm] = cat8(*(const v8bf*)(base[mm] + off),
                       *(const v8bf*)(base[mm] + off + 16));
        int ktg = s * 2 + kk;             // global 32-wide K tile (of 18)
#pragma unroll
        for (int u = 0; u < 2; ++u) {
          v16bf b = *(const v16bf*)(sB + (((ktg * 2 + u) * 32 + lane) * 16));
#pragma unroll
          for (int mm = 0; mm < 2; ++mm)
            acc[mm][u] = __builtin_amdgcn_wmma_f32_16x16x32_bf16(false, a[mm], false, b,
                                                                 (short)0, acc[mm][u],
                                                                 false, false);
        }
      }
    }
  }
#pragma unroll
  for (int u = 0; u < 2; ++u) {
    int col = u * 16 + (lane & 15);
    if (col < 25) {
      float bv = bias[col];
#pragma unroll
      for (int mm = 0; mm < 2; ++mm)
#pragma unroll
        for (int v = 0; v < 8; ++v) {
          int r = m0 + mm * 16 + v + hi * 8;
          out[(size_t)r * 25 + col] = acc[mm][u][v] + bv;
        }
    }
  }
}

// ---------------------------------------------------------------------------
// CARAFE: one block per COARSE pixel; the 4 fine pixels of its 2x2 block
// share the same 25-tap patch, so each tap (256 ch) is loaded once and
// reused under 4 softmax masks -> 4x less gather traffic.
// ---------------------------------------------------------------------------
__global__ void k_carafe(const float* __restrict__ fcont, const float* __restrict__ ccont,
                         const float* __restrict__ coarse, __bf16* __restrict__ reass) {
  __shared__ float slog[4][25];
  __shared__ float sexp[4][25];
  int cp = blockIdx.x;                // coarse pixel, B*h*w = 16384
  int tid = threadIdx.x;
  int b = cp >> 12;
  int remc = cp & 4095;
  int ci = remc >> 6, cj = remc & 63;
  if (tid < 100) {
    int q = tid / 25, t = tid % 25;
    int i = 2 * ci + (q >> 1), j = 2 * cj + (q & 1);
    int p = (b * 128 + i) * 128 + j;
    slog[q][t] = fcont[(size_t)p * 25 + t] + ccont[(size_t)cp * 25 + t];
  }
  __syncthreads();
  if (tid < 100) {
    int q = tid / 25, t = tid % 25;
    float mx = slog[q][0];
    for (int tt = 1; tt < 25; ++tt) mx = fmaxf(mx, slog[q][tt]);
    sexp[q][t] = __expf(slog[q][t] - mx);
  }
  __syncthreads();
  float inv[4];
#pragma unroll
  for (int q = 0; q < 4; ++q) {
    float sum = 0.f;
    for (int t = 0; t < 25; ++t) sum += sexp[q][t];
    inv[q] = 1.0f / sum;
  }
  int c = tid;                        // channel 0..255
  float acc[4] = {0.f, 0.f, 0.f, 0.f};
#pragma unroll
  for (int t = 0; t < 25; ++t) {
    int dy = t / 5 - 2, dx = t % 5 - 2;
    int yy = ci + dy, xx = cj + dx;
    if (yy >= 0 && yy < 64 && xx >= 0 && xx < 64) {
      float val = coarse[(((size_t)(b * 64 + yy)) * 64 + xx) * 256 + c];
#pragma unroll
      for (int q = 0; q < 4; ++q) acc[q] += sexp[q][t] * val;
    }
  }
#pragma unroll
  for (int q = 0; q < 4; ++q) {
    int i = 2 * ci + (q >> 1), j = 2 * cj + (q & 1);
    int p = (b * 128 + i) * 128 + j;
    reass[(size_t)p * 256 + c] = (__bf16)(acc[q] * inv[q]);
  }
}

// ---------------------------------------------------------------------------
// Final GEMM fused with gated blend, MB=2 x NB=4, LDS-staged B:
//   fine_o = coarse_o(bf16) * fine_w + fine_b     (WMMA)
//   out    = gate*fine_o + (1-gate)*coarse_o      (fp32)
// M=65536, N=K=256; grid = (M/256, 4).
// ---------------------------------------------------------------------------
__global__ void k_gemm_final(const __bf16* __restrict__ A, const __bf16* __restrict__ Bp,
                             const float* __restrict__ bias, const float* __restrict__ gate,
                             float* __restrict__ out) {
  const int N = 256, K = 256, KT = 8, NB = 4;
  __shared__ __bf16 sB[KT * NB * 32 * 16];
  int lane = threadIdx.x & 31;
  int wave = threadIdx.x >> 5;
  int nt0 = blockIdx.y * NB;
  stage_b<KT, NB>(Bp, sB, N >> 4, nt0);

  int m0 = (blockIdx.x * WAVES + wave) * 32;
  int hi = lane >> 4;
  v8f acc[2][NB] = {};
  for (int kt = 0; kt < KT; ++kt) {
    v16bf a[2];
#pragma unroll
    for (int mm = 0; mm < 2; ++mm) {
      int row = m0 + mm * 16 + (lane & 15);
      const __bf16* ap = A + (size_t)row * K + kt * 32 + hi * 8;
      if (kt + 1 < KT) __builtin_prefetch(ap + 32, 0, 0);
      a[mm] = cat8(*(const v8bf*)ap, *(const v8bf*)(ap + 16));
    }
#pragma unroll
    for (int u = 0; u < NB; ++u) {
      v16bf b = *(const v16bf*)(sB + (((kt * NB + u) * 32 + lane) * 16));
#pragma unroll
      for (int mm = 0; mm < 2; ++mm)
        acc[mm][u] = __builtin_amdgcn_wmma_f32_16x16x32_bf16(false, a[mm], false, b,
                                                             (short)0, acc[mm][u],
                                                             false, false);
    }
  }
#pragma unroll
  for (int u = 0; u < NB; ++u) {
    int col = (nt0 + u) * 16 + (lane & 15);
    float bv = bias[col];
#pragma unroll
    for (int mm = 0; mm < 2; ++mm)
#pragma unroll
      for (int v = 0; v < 8; ++v) {
        int r = m0 + mm * 16 + v + hi * 8;
        int b = r >> 14;
        int rem = r & 16383;
        int i = rem >> 7, j = rem & 127;
        float g = gate[(b * 64 + (i >> 1)) * 64 + (j >> 1)];
        float co = (float)A[(size_t)r * K + col];    // coarse_o (K==N==256)
        float fo = acc[mm][u][v] + bv;
        out[(size_t)r * N + col] = g * fo + (1.0f - g) * co;
      }
  }
}

// ---------------------------------------------------------------------------
extern "C" void kernel_launch(void* const* d_in, const int* in_sizes, int n_in,
                              void* d_out, int out_size, void* d_ws, size_t ws_size,
                              hipStream_t stream) {
  (void)in_sizes; (void)n_in; (void)out_size; (void)ws_size;
  const float* fine      = (const float*)d_in[0];
  const float* coarse    = (const float*)d_in[1];
  const float* gate_w    = (const float*)d_in[2];
  const float* gate_b    = (const float*)d_in[3];
  const float* sfine_w   = (const float*)d_in[4];
  const float* sfine_b   = (const float*)d_in[5];
  const float* scoarse_w = (const float*)d_in[6];
  const float* content_w = (const float*)d_in[7];
  const float* content_b = (const float*)d_in[8];
  const float* coarse_w  = (const float*)d_in[9];
  const float* coarse_b  = (const float*)d_in[10];
  const float* fine_w    = (const float*)d_in[11];
  const float* fine_b    = (const float*)d_in[12];
  float* out = (float*)d_out;

  char* ws = (char*)d_ws;
  size_t off = 0;
  auto alloc = [&](size_t bytes) -> void* {
    void* p = ws + off;
    off = (off + bytes + 255) & ~(size_t)255;
    return p;
  };
  const int Mf = 4 * 128 * 128;   // 65536 fine pixels
  const int Mc = 4 * 64 * 64;     // 16384 coarse pixels
  const size_t fpad_elems = (size_t)4 * 130 * 130 * 64;   // padded fine emb
  const size_t cpad_elems = (size_t)4 * 66 * 66 * 64;     // padded coarse emb

  __bf16* pk_sfine   = (__bf16*)alloc((size_t)256 * 64 * 2);
  __bf16* pk_scoarse = (__bf16*)alloc((size_t)256 * 64 * 2);
  __bf16* pk_content = (__bf16*)alloc((size_t)576 * 32 * 2);
  __bf16* pk_cw      = (__bf16*)alloc((size_t)256 * 256 * 2);
  __bf16* pk_fw      = (__bf16*)alloc((size_t)256 * 256 * 2);
  float*  gate       = (float*) alloc((size_t)Mc * 4);
  __bf16* femb       = (__bf16*)alloc(fpad_elems * 2);
  __bf16* cemb       = (__bf16*)alloc(cpad_elems * 2);
  float*  fcont      = (float*) alloc((size_t)Mf * 25 * 4);
  float*  ccont      = (float*) alloc((size_t)Mc * 25 * 4);
  __bf16* reass      = (__bf16*)alloc((size_t)Mf * 256 * 2);
  __bf16* co         = (__bf16*)alloc((size_t)Mf * 256 * 2);

  // Zero the padded embedding buffers (halo must be 0 for SAME conv).
  hipMemsetAsync(femb, 0, fpad_elems * 2, stream);
  hipMemsetAsync(cemb, 0, cpad_elems * 2, stream);

  // Pack weights into WMMA-B fragment order.
  k_pack_b<<<(256 * 64 + 255) / 256, 256, 0, stream>>>(sfine_w,   pk_sfine,   256, 64, 64);
  k_pack_b<<<(256 * 64 + 255) / 256, 256, 0, stream>>>(scoarse_w, pk_scoarse, 256, 64, 64);
  k_pack_b<<<(576 * 32 + 255) / 256, 256, 0, stream>>>(content_w, pk_content, 576, 25, 32);
  k_pack_b<<<(256 * 256 + 255) / 256, 256, 0, stream>>>(coarse_w, pk_cw,      256, 256, 256);
  k_pack_b<<<(256 * 256 + 255) / 256, 256, 0, stream>>>(fine_w,   pk_fw,      256, 256, 256);

  // Gate at coarse resolution.
  k_gate<<<Mc / 8, 256, 0, stream>>>(coarse, gate_w, gate_b, gate, Mc);

  // Embeddings (WMMA GEMMs), stored into zero-halo padded NHWC layout.
  k_gemm_a32<8, 4, 128, 128><<<dim3(Mf / 256, 1), 256, 0, stream>>>(
      fine, pk_sfine, sfine_b, femb, 64);
  k_gemm_a32<8, 4, 64, 64><<<dim3(Mc / 256, 1), 256, 0, stream>>>(
      coarse, pk_scoarse, nullptr, cemb, 64);

  // Content 3x3 convs: branch-free shifted WMMA over padded embeddings.
  k_conv3x3<128, 128><<<dim3(Mf / 256, 1), 256, 0, stream>>>(femb, pk_content,
                                                             content_b, fcont);
  k_conv3x3<64, 64><<<dim3(Mc / 256, 1), 256, 0, stream>>>(cemb, pk_content,
                                                           content_b, ccont);

  // Softmax + content-aware reassembly (one block per coarse pixel).
  k_carafe<<<Mc, 256, 0, stream>>>(fcont, ccont, coarse, reass);

  // coarse_o = reass * coarse_w + coarse_b  (WMMA, MB=2 x NB=4, LDS B).
  k_gemm_a16<8, 4><<<dim3(Mf / 256, 4), 256, 0, stream>>>(reass, pk_cw, coarse_b,
                                                          co, 256);

  // fine_o GEMM fused with gated blend -> output.
  k_gemm_final<<<dim3(Mf / 256, 4), 256, 0, stream>>>(co, pk_fw, fine_b, gate, out);
}
